// TransformerLinear_34488587386970
// MI455X (gfx1250) — matrix operand order repaired and checked by reading, not compiled, not run on hardware
//
#include <hip/hip_runtime.h>
#include <math.h>

#define HID    32
#define HEADS  2
#define HDIM   16
#define LAYERS 12
#define ZDIM   16
#define IN_F   1024
#define OUT_F  1024
#define NTOK   512
#define NROWS  65536
#define EPS    1e-5f

typedef __attribute__((ext_vector_type(16))) __bf16 v16bf;
typedef __attribute__((ext_vector_type(8)))  __bf16 v8bf;
typedef __attribute__((ext_vector_type(8)))  float  v8f;

union V16U { v16bf v; v8bf h[2]; };

static __device__ __forceinline__ unsigned short f2bf(float f) {
    unsigned int u = __float_as_uint(f);
    u += 0x7FFFu + ((u >> 16) & 1u);          // round-to-nearest-even
    return (unsigned short)(u >> 16);
}

static __device__ __forceinline__ float wave_sum32(float v) {
    #pragma unroll
    for (int off = 16; off > 0; off >>= 1) v += __shfl_xor(v, off, 32);
    return v;
}

// ---------------------------------------------------------------- K0: embed
__global__ void k_embed(const float* __restrict__ z, const float* __restrict__ tw,
                        float* __restrict__ h) {
    int t = blockIdx.x, c = threadIdx.x;          // 512 x 32
    float s = 0.f;
    #pragma unroll
    for (int j = 0; j < ZDIM; ++j) s += z[t * ZDIM + j] * tw[j * HID + c];
    h[t * HID + c] = s;
}

// ------------------------------------------------------ K1: LN1 + QKV (per layer)
__global__ void k_qkv(const float* __restrict__ h,
                      const float* __restrict__ g1, const float* __restrict__ b1,
                      const float* __restrict__ wq, const float* __restrict__ wk,
                      const float* __restrict__ wv,
                      float* __restrict__ qb, float* __restrict__ kb, float* __restrict__ vb) {
    int t = blockIdx.x, c = threadIdx.x;          // 512 blocks x 32 threads (1 wave)
    float hv = h[t * HID + c];
    float mean = wave_sum32(hv) * (1.f / HID);
    float d = hv - mean;
    float var = wave_sum32(d * d) * (1.f / HID);
    float hn = d * rsqrtf(var + EPS) * g1[c] + b1[c];
    float q = 0.f, k = 0.f, v = 0.f;
    #pragma unroll
    for (int j = 0; j < HID; ++j) {
        float hj = __shfl(hn, j, 32);
        q += hj * wq[j * HID + c];
        k += hj * wk[j * HID + c];
        v += hj * wv[j * HID + c];
    }
    qb[t * HID + c] = q; kb[t * HID + c] = k; vb[t * HID + c] = v;
}

// ---------------- K2: attention + Wo + residual + LN2 + FFN + residual (per layer)
__global__ void k_attn(float* __restrict__ h,
                       const float* __restrict__ qb, const float* __restrict__ kb,
                       const float* __restrict__ vb,
                       const float* __restrict__ wo, const float* __restrict__ bo,
                       const float* __restrict__ g2, const float* __restrict__ b2,
                       const float* __restrict__ w1, const float* __restrict__ w2) {
    __shared__ float ot[2][HID];
    __shared__ float hrow[2][HID];
    __shared__ float hn2[2][HID];
    __shared__ float a1[2][4 * HID];

    int tid = threadIdx.x;                 // 128 threads = 4 waves
    int wid = tid >> 5, lane = tid & 31;
    int r = wid >> 1, head = wid & 1;
    int t = blockIdx.x * 2 + r;

    // flash attention: one wave per (row, head), lanes split K tokens
    float qd[HDIM];
    #pragma unroll
    for (int d = 0; d < HDIM; ++d) qd[d] = qb[t * HID + head * HDIM + d];
    float m = -INFINITY, l = 0.f, acc[HDIM];
    #pragma unroll
    for (int d = 0; d < HDIM; ++d) acc[d] = 0.f;
    #pragma unroll 4
    for (int i = 0; i < NTOK / 32; ++i) {
        int kidx = i * 32 + lane;
        const float* kr = kb + kidx * HID + head * HDIM;
        const float* vr = vb + kidx * HID + head * HDIM;
        float s = 0.f;
        #pragma unroll
        for (int d = 0; d < HDIM; ++d) s += qd[d] * kr[d];
        s *= 0.25f;                                       // 1/sqrt(HDIM)
        float mN = fmaxf(m, s);
        float corr = __expf(m - mN), e = __expf(s - mN);
        l = l * corr + e;
        #pragma unroll
        for (int d = 0; d < HDIM; ++d) acc[d] = acc[d] * corr + e * vr[d];
        m = mN;
    }
    #pragma unroll
    for (int off = 16; off > 0; off >>= 1) {
        float m2 = __shfl_xor(m, off, 32), l2 = __shfl_xor(l, off, 32);
        float M = fmaxf(m, m2);
        float c1 = __expf(m - M), c2 = __expf(m2 - M);
        l = l * c1 + l2 * c2;
        #pragma unroll
        for (int d = 0; d < HDIM; ++d) {
            float a2 = __shfl_xor(acc[d], off, 32);
            acc[d] = acc[d] * c1 + a2 * c2;
        }
        m = M;
    }
    if (lane == 0) {
        float inv = 1.f / l;
        #pragma unroll
        for (int d = 0; d < HDIM; ++d) ot[r][head * HDIM + d] = acc[d] * inv;
    }
    __syncthreads();

    // h += o @ wo + bo
    if (tid < 64) {
        int rr = tid >> 5, c = tid & 31;
        int t2 = blockIdx.x * 2 + rr;
        float s = h[t2 * HID + c] + bo[c];
        #pragma unroll
        for (int j = 0; j < HID; ++j) s += ot[rr][j] * wo[j * HID + c];
        hrow[rr][c] = s;
    }
    __syncthreads();

    // LN2 (wave0 -> row0, wave1 -> row1)
    if (wid < 2) {
        int rr = wid, c = lane;
        float hv = hrow[rr][c];
        float mean = wave_sum32(hv) * (1.f / HID);
        float d = hv - mean;
        float var = wave_sum32(d * d) * (1.f / HID);
        hn2[rr][c] = d * rsqrtf(var + EPS) * g2[c] + b2[c];
    }
    __syncthreads();

    // FFN up + exact gelu
    for (int idx = tid; idx < 2 * 4 * HID; idx += 128) {
        int rr = idx >> 7, j = idx & 127;
        float x = 0.f;
        #pragma unroll
        for (int c = 0; c < HID; ++c) x += hn2[rr][c] * w1[c * (4 * HID) + j];
        a1[rr][j] = 0.5f * x * (1.f + erff(x * 0.70710678118654752f));
    }
    __syncthreads();

    // FFN down + residual
    if (tid < 64) {
        int rr = tid >> 5, c = tid & 31;
        int t2 = blockIdx.x * 2 + rr;
        float s = hrow[rr][c];
        #pragma unroll
        for (int j = 0; j < 4 * HID; ++j) s += a1[rr][j] * w2[j * HID + c];
        h[t2 * HID + c] = s;
    }
}

// ------------------------------------------------------------- K3a: final LN
__global__ void k_fln(const float* __restrict__ h, const float* __restrict__ g,
                      const float* __restrict__ b, float* __restrict__ hf) {
    int t = blockIdx.x, c = threadIdx.x;
    float hv = h[t * HID + c];
    float mean = wave_sum32(hv) * (1.f / HID);
    float d = hv - mean;
    float var = wave_sum32(d * d) * (1.f / HID);
    hf[t * HID + c] = d * rsqrtf(var + EPS) * g[c] + b[c];
}

// ------------------- K3b: key/value projections, packed bf16 (V transposed)
__global__ void k_kv(const float* __restrict__ hf,
                     const float* __restrict__ kw, const float* __restrict__ kbias,
                     const float* __restrict__ vw, const float* __restrict__ vbias,
                     unsigned short* __restrict__ Kbf, unsigned short* __restrict__ Vt) {
    int gid = blockIdx.x * blockDim.x + threadIdx.x;  // 2*512*1024 threads
    if (gid < NTOK * IN_F) {
        int t = gid / IN_F, f = gid % IN_F;
        float s = kbias[f];
        #pragma unroll
        for (int c = 0; c < HID; ++c) s += hf[t * HID + c] * kw[c * IN_F + f];
        Kbf[t * IN_F + f] = f2bf(s);
    } else {
        int g2 = gid - NTOK * IN_F;
        int t = g2 / OUT_F, f = g2 % OUT_F;
        float s = vbias[f];
        #pragma unroll
        for (int c = 0; c < HID; ++c) s += hf[t * HID + c] * vw[c * OUT_F + f];
        Vt[f * NTOK + t] = f2bf(s);                   // transposed: Vt[n][t]
    }
}

// ---------------- K4: fused  softmax(x @ K^T / 32) @ V  via bf16 WMMA
__global__ void __launch_bounds__(256)
k_flash(const float* __restrict__ x, const unsigned short* __restrict__ Kbf,
        const unsigned short* __restrict__ Vt, float* __restrict__ out) {
    __shared__ unsigned short Xl[16 * IN_F];   // 32 KB, x tile in bf16
    __shared__ unsigned short Pl[16 * NTOK];   // 16 KB, softmax probs in bf16
    __shared__ float red[8][16];
    __shared__ float rowmax[16], rowsum[16];

    const int tid = threadIdx.x;               // 256 threads = 8 waves
    const int wid = tid >> 5, lane = tid & 31;
    const int hi = lane >> 4, ln = lane & 15;
    const int row0 = blockIdx.x * 16;

    // stage x tile -> LDS as bf16
    for (int idx = tid; idx < 16 * IN_F; idx += 256)
        Xl[idx] = f2bf(x[(row0 + (idx >> 10)) * IN_F + (idx & 1023)]);
    __syncthreads();

    // ---- phase 1: S[16, wave's 64 cols] = Xtile @ K^T   (K-dim = 1024)
    v8f s[4];
    #pragma unroll
    for (int t = 0; t < 4; ++t) {
        #pragma unroll
        for (int e = 0; e < 8; ++e) s[t][e] = 0.f;
    }
    for (int kk = 0; kk < IN_F / 32; ++kk) {
        const int kbase = kk * 32;
        V16U a;
        a.h[0] = *reinterpret_cast<const v8bf*>(&Xl[ln * IN_F + kbase + 16 * hi]);
        a.h[1] = *reinterpret_cast<const v8bf*>(&Xl[ln * IN_F + kbase + 16 * hi + 8]);
        #pragma unroll
        for (int t = 0; t < 4; ++t) {
            const int tok = wid * 64 + t * 16 + ln;
            v16bf b = *reinterpret_cast<const v16bf*>(&Kbf[tok * IN_F + kbase + 16 * hi]);
            s[t] = __builtin_amdgcn_wmma_f32_16x16x32_bf16(
                false, a.v, false, b, (short)0, s[t], false, false);
        }
    }
    #pragma unroll
    for (int t = 0; t < 4; ++t) {
        #pragma unroll
        for (int e = 0; e < 8; ++e) s[t][e] *= (1.f / 32.f);   // /sqrt(IN_F)
    }

    // ---- softmax statistics: max
    float rm[8];
    #pragma unroll
    for (int r = 0; r < 8; ++r) {
        rm[r] = s[0][r];
        #pragma unroll
        for (int t = 1; t < 4; ++t) rm[r] = fmaxf(rm[r], s[t][r]);
        #pragma unroll
        for (int off = 1; off < 16; off <<= 1) rm[r] = fmaxf(rm[r], __shfl_xor(rm[r], off, 32));
    }
    if (ln == 0) {
        #pragma unroll
        for (int r = 0; r < 8; ++r) red[wid][r + 8 * hi] = rm[r];
    }
    __syncthreads();
    if (tid < 16) {
        float m = -INFINITY;
        #pragma unroll
        for (int w = 0; w < 8; ++w) m = fmaxf(m, red[w][tid]);
        rowmax[tid] = m;
    }
    __syncthreads();

    // ---- exp + sum
    float ls[8];
    #pragma unroll
    for (int r = 0; r < 8; ++r) {
        float mx = rowmax[r + 8 * hi];
        ls[r] = 0.f;
        #pragma unroll
        for (int t = 0; t < 4; ++t) {
            float e = __expf(s[t][r] - mx);
            s[t][r] = e;
            ls[r] += e;
        }
        #pragma unroll
        for (int off = 1; off < 16; off <<= 1) ls[r] += __shfl_xor(ls[r], off, 32);
    }
    if (ln == 0) {
        #pragma unroll
        for (int r = 0; r < 8; ++r) red[wid][r + 8 * hi] = ls[r];
    }
    __syncthreads();
    if (tid < 16) {
        float sm = 0.f;
        #pragma unroll
        for (int w = 0; w < 8; ++w) sm += red[w][tid];
        rowsum[tid] = sm;
    }
    __syncthreads();

    // ---- write normalized probs to LDS as bf16
    #pragma unroll
    for (int r = 0; r < 8; ++r) {
        float inv = 1.f / rowsum[r + 8 * hi];
        #pragma unroll
        for (int t = 0; t < 4; ++t)
            Pl[(r + 8 * hi) * NTOK + wid * 64 + t * 16 + ln] = f2bf(s[t][r] * inv);
    }
    __syncthreads();

    // ---- phase 2: O[16, wave's 128 cols] = P @ V   (K-dim = 512)
    v8f o[8];
    #pragma unroll
    for (int nt = 0; nt < 8; ++nt) {
        #pragma unroll
        for (int e = 0; e < 8; ++e) o[nt][e] = 0.f;
    }
    const int n0 = wid * 128;
    for (int kk = 0; kk < NTOK / 32; ++kk) {
        const int kbase = kk * 32;
        V16U a;
        a.h[0] = *reinterpret_cast<const v8bf*>(&Pl[ln * NTOK + kbase + 16 * hi]);
        a.h[1] = *reinterpret_cast<const v8bf*>(&Pl[ln * NTOK + kbase + 16 * hi + 8]);
        #pragma unroll
        for (int nt = 0; nt < 8; ++nt) {
            const int n = n0 + nt * 16 + ln;
            v16bf b = *reinterpret_cast<const v16bf*>(&Vt[n * NTOK + kbase + 16 * hi]);
            o[nt] = __builtin_amdgcn_wmma_f32_16x16x32_bf16(
                false, a.v, false, b, (short)0, o[nt], false, false);
        }
    }
    #pragma unroll
    for (int nt = 0; nt < 8; ++nt) {
        #pragma unroll
        for (int r = 0; r < 8; ++r)
            out[(row0 + r + 8 * hi) * OUT_F + n0 + nt * 16 + ln] = o[nt][r];
    }
}

// ---------------------------------------------------------------- launcher
extern "C" void kernel_launch(void* const* d_in, const int* in_sizes, int n_in,
                              void* d_out, int out_size, void* d_ws, size_t ws_size,
                              hipStream_t stream) {
    const float* x       = (const float*)d_in[0];
    const float* z_tok   = (const float*)d_in[1];
    const float* token_w = (const float*)d_in[2];
    const float* wq      = (const float*)d_in[3];
    const float* wk      = (const float*)d_in[4];
    const float* wv      = (const float*)d_in[5];
    const float* wo      = (const float*)d_in[6];
    const float* bo      = (const float*)d_in[7];
    const float* ln1_g   = (const float*)d_in[8];
    const float* ln1_b   = (const float*)d_in[9];
    const float* ln2_g   = (const float*)d_in[10];
    const float* ln2_b   = (const float*)d_in[11];
    const float* w1      = (const float*)d_in[12];
    const float* w2      = (const float*)d_in[13];
    const float* fln_g   = (const float*)d_in[14];
    const float* fln_b   = (const float*)d_in[15];
    const float* key_w   = (const float*)d_in[16];
    const float* key_b   = (const float*)d_in[17];
    const float* value_w = (const float*)d_in[18];
    const float* value_b = (const float*)d_in[19];
    float* out = (float*)d_out;

    char* ws = (char*)d_ws;
    float* h  = (float*)(ws);                               // 512*32 f32
    float* qb = (float*)(ws + 65536);
    float* kb = (float*)(ws + 131072);
    float* vb = (float*)(ws + 196608);
    float* hf = (float*)(ws + 262144);
    unsigned short* Kbf = (unsigned short*)(ws + 327680);   // 512*1024 bf16
    unsigned short* Vt  = (unsigned short*)(ws + 327680 + NTOK * IN_F * 2);

    k_embed<<<NTOK, HID, 0, stream>>>(z_tok, token_w, h);

    for (int l = 0; l < LAYERS; ++l) {
        const float* wq_l = wq + (size_t)l * HID * HID;
        const float* wk_l = wk + (size_t)l * HID * HID;
        const float* wv_l = wv + (size_t)l * HID * HID;
        const float* wo_l = wo + (size_t)l * HID * HID;
        const float* bo_l = bo + (size_t)l * HID;
        const float* g1_l = ln1_g + (size_t)l * HID;
        const float* b1_l = ln1_b + (size_t)l * HID;
        const float* g2_l = ln2_g + (size_t)l * HID;
        const float* b2_l = ln2_b + (size_t)l * HID;
        const float* w1_l = w1 + (size_t)l * HID * 4 * HID;
        const float* w2_l = w2 + (size_t)l * 4 * HID * HID;
        k_qkv<<<NTOK, HID, 0, stream>>>(h, g1_l, b1_l, wq_l, wk_l, wv_l, qb, kb, vb);
        k_attn<<<NTOK / 2, 128, 0, stream>>>(h, qb, kb, vb, wo_l, bo_l, g2_l, b2_l, w1_l, w2_l);
    }

    k_fln<<<NTOK, HID, 0, stream>>>(h, fln_g, fln_b, hf);
    k_kv<<<(2 * NTOK * IN_F) / 256, 256, 0, stream>>>(hf, key_w, key_b, value_w, value_b, Kbf, Vt);
    k_flash<<<NROWS / 16, 256, 0, stream>>>(x, Kbf, Vt, out);
}
